// TimeDecoder_25503515803677
// MI455X (gfx1250) — matrix operand order repaired and checked by reading, not compiled, not run on hardware
//
#include <hip/hip_runtime.h>
#include <cstddef>

// ---------------- model constants ----------------
#define Cc   128
#define Hh   8
#define DHd  16
#define DIi  256
#define DSs  16
#define DRr  8
#define DCc  4
#define FFf  512
#define L1l  4
#define L2l  4
#define Llay 8
#define Bb   32
#define Mm   6
#define Tt   60
#define Nn   512
#define TOK  (Mm*Tt)          // 360
#define ROWS (Bb*TOK)         // 11520
#define SEQ  (Bb*Mm)          // 192
#define ENCR (Bb*Nn)          // 16384

typedef __attribute__((ext_vector_type(16))) __bf16 bf16x16;
typedef __attribute__((ext_vector_type(8)))  float  f32x8;

// k-index swizzle: swap bits 3 and 4 so one WMMA A-fragment is 16 contiguous bf16
__device__ __forceinline__ int swz(int k) {
  return (k & ~0x18) | ((k & 8) << 1) | ((k & 16) >> 1);
}

__device__ __forceinline__ float gelu_f(float x){
  float x3 = x*x*x;
  return 0.5f*x*(1.0f + tanhf(0.7978845608028654f*(x + 0.044715f*x3)));
}
__device__ __forceinline__ float silu_f(float x){ return x/(1.0f + expf(-x)); }
__device__ __forceinline__ float softplus_f(float x){ return (x > 20.0f) ? x : log1pf(expf(x)); }

// ============ one-time weight transpose + bf16 pack: Wt[l][n][k] = W[l][k][n] ============
__global__ __launch_bounds__(256)
void wconv_k(const float* __restrict__ W, __bf16* __restrict__ Wt,
             int total, int K, int N)
{
  int gid = blockIdx.x * blockDim.x + threadIdx.x;
  if (gid >= total) return;
  int l = gid / (K*N);
  int r = gid - l*(K*N);
  int k = r / N;
  int n = r - k*N;
  Wt[(size_t)l*N*K + (size_t)n*K + swz(k)] = (__bf16)W[gid];
}

// ============ pack encoding to bf16 swizzled (A-side of K/V projections) ============
__global__ __launch_bounds__(256)
void encpack_k(const float* __restrict__ E, __bf16* __restrict__ Eb)
{
  int gid = blockIdx.x * blockDim.x + threadIdx.x;
  if (gid >= ENCR*Cc) return;
  int c = gid & (Cc-1);
  Eb[(gid & ~(Cc-1)) + swz(c)] = (__bf16)E[gid];
}

// ============ pack K (bf16 rowmajor) and V (bf16 transposed per head) ============
__global__ __launch_bounds__(256)
void kvpack_k(const float* __restrict__ kf32, const float* __restrict__ vf32,
              __bf16* __restrict__ kbf, __bf16* __restrict__ vtb)
{
  int gid = blockIdx.x * blockDim.x + threadIdx.x;
  if (gid >= ENCR*Cc) return;
  kbf[gid] = (__bf16)kf32[gid];
  int col = gid & (Cc-1);
  int row = gid >> 7;          // b*Nn + key
  int b   = row >> 9;
  int key = row & (Nn-1);
  vtb[((size_t)b*Cc + col)*Nn + key] = (__bf16)vf32[gid];
}

// =====================================================================
// WMMA GEMM, A bf16-swizzled [rows][K], B bf16 transposed [N][K].
// rows%64==0, K%32==0, ragged N ok. Inner loop: 5x32B loads + 4 WMMA.
// Output: f32 (+bias/act/resid) or bf16-swizzled (+bias/act).
// act: 0 none, 1 gelu, 2 softplus
// =====================================================================
__global__ __launch_bounds__(32)
void gemm_bb_k(const __bf16* __restrict__ Ab,
               const __bf16* __restrict__ Wt,
               const float* __restrict__ bias,
               const float* __restrict__ resid,
               float* __restrict__ out,
               __bf16* __restrict__ outb,
               int ldo, int K, int N, int act)
{
  const int lane = threadIdx.x;
  const int half = lane >> 4, l16 = lane & 15;
  const int m0 = blockIdx.x * 64, n0 = blockIdx.y * 16;
  const int bcol = n0 + l16;
  const int bc = (bcol < N) ? bcol : (N - 1);   // clamp; extra cols never stored
  f32x8 acc0 = {}, acc1 = {}, acc2 = {}, acc3 = {};
  const __bf16* Wr = Wt + (size_t)bc*K + half*16;
  const __bf16* A0 = Ab + (size_t)(m0 +  0 + l16)*K + half*16;
  const __bf16* A1 = Ab + (size_t)(m0 + 16 + l16)*K + half*16;
  const __bf16* A2 = Ab + (size_t)(m0 + 32 + l16)*K + half*16;
  const __bf16* A3 = Ab + (size_t)(m0 + 48 + l16)*K + half*16;
  for (int k0 = 0; k0 < K; k0 += 32) {
    const bf16x16 bfv = *(const bf16x16*)(Wr + k0);
    const bf16x16 af0 = *(const bf16x16*)(A0 + k0);
    const bf16x16 af1 = *(const bf16x16*)(A1 + k0);
    const bf16x16 af2 = *(const bf16x16*)(A2 + k0);
    const bf16x16 af3 = *(const bf16x16*)(A3 + k0);
    acc0 = __builtin_amdgcn_wmma_f32_16x16x32_bf16(false, af0, false, bfv, (short)0, acc0, false, false);
    acc1 = __builtin_amdgcn_wmma_f32_16x16x32_bf16(false, af1, false, bfv, (short)0, acc1, false, false);
    acc2 = __builtin_amdgcn_wmma_f32_16x16x32_bf16(false, af2, false, bfv, (short)0, acc2, false, false);
    acc3 = __builtin_amdgcn_wmma_f32_16x16x32_bf16(false, af3, false, bfv, (short)0, acc3, false, false);
  }
  if (bcol < N) {
    const float bb = bias ? bias[bcol] : 0.f;
    const int sb = swz(bcol);
#pragma unroll
    for (int mt = 0; mt < 4; ++mt) {
      f32x8 acc = (mt == 0) ? acc0 : (mt == 1) ? acc1 : (mt == 2) ? acc2 : acc3;
#pragma unroll
      for (int v = 0; v < 8; ++v) {
        int m = m0 + mt*16 + half*8 + v;     // C/D layout: m = v + 8*half
        float val = acc[v] + bb;
        if (act == 1)      val = gelu_f(val);
        else if (act == 2) val = softplus_f(val);
        if (outb) {
          outb[(size_t)m*ldo + sb] = (__bf16)val;
        } else {
          if (resid) val += resid[(size_t)m*ldo + bcol];
          out[(size_t)m*ldo + bcol] = val;
        }
      }
    }
  }
}

// =====================================================================
// EDGE WMMA GEMM (only user: dt-proj, K=8, f32 A): clamp+select.
// =====================================================================
__global__ __launch_bounds__(32)
void gemm_edge_k(const float* __restrict__ A, int lda,
                 const float* __restrict__ W, int ldw,
                 const float* __restrict__ bias,
                 float* __restrict__ out, int ldo,
                 int K, int N, int act)
{
  const int lane = threadIdx.x;
  const int half = lane >> 4, l16 = lane & 15;
  const int m0 = blockIdx.x * 16, n0 = blockIdx.y * 16;
  const int bcol = n0 + l16;
  const int bc = (bcol < N) ? bcol : 0;
  f32x8 acc = {};
  const int arow = m0 + l16;
  for (int k0 = 0; k0 < K; k0 += 32) {
    bf16x16 af, bfv;
#pragma unroll
    for (int v = 0; v < 8; ++v) {
      int ka = k0 + (v >> 2)*16 + half*8 + (v & 3)*2;
      float a0 = A[(size_t)arow*lda + ((ka     < K) ? ka     : 0)];
      float a1 = A[(size_t)arow*lda + ((ka + 1 < K) ? ka + 1 : 0)];
      af[2*v]   = (__bf16)((ka     < K) ? a0 : 0.f);
      af[2*v+1] = (__bf16)((ka + 1 < K) ? a1 : 0.f);
      int kb = k0 + (v >> 2)*16 + half*8 + (v & 3)*2;
      float b0 = W[(size_t)((kb     < K) ? kb     : 0)*ldw + bc];
      float b1 = W[(size_t)((kb + 1 < K) ? kb + 1 : 0)*ldw + bc];
      bfv[2*v]   = (__bf16)((kb     < K) ? b0 : 0.f);
      bfv[2*v+1] = (__bf16)((kb + 1 < K) ? b1 : 0.f);
    }
    acc = __builtin_amdgcn_wmma_f32_16x16x32_bf16(false, af, false, bfv,
                                                  (short)0, acc, false, false);
  }
  if (bcol < N) {
    float bb = bias ? bias[bcol] : 0.f;
#pragma unroll
    for (int v = 0; v < 8; ++v) {
      int m = m0 + half*8 + v;
      float val = acc[v] + bb;
      if (act == 1)      val = gelu_f(val);
      else if (act == 2) val = softplus_f(val);
      out[(size_t)m*ldo + bcol] = val;
    }
  }
}

// =====================================================================
// Fused cross-attention: bf16 K/V packs, bf16-swizzled output + LDS probs.
// =====================================================================
__global__ __launch_bounds__(32)
void attn_wmma_k(const float* __restrict__ Q, const __bf16* __restrict__ Kbf,
                 const __bf16* __restrict__ Vtb, const unsigned char* __restrict__ mask,
                 __bf16* __restrict__ Ob)
{
  __shared__ __align__(32) float  sS[16 * Nn];   // 32 KB scores/exp
  __shared__ __align__(32) __bf16 sP[16 * Nn];   // 16 KB probs (row-major bf16)
  const int lane = threadIdx.x, half = lane >> 4, l16 = lane & 15;
  const int m0 = blockIdx.x * 16;
  const int h  = blockIdx.y;
  const int b  = blockIdx.z;
  const int arow = m0 + l16;
  const int ar   = (arow < TOK) ? arow : (TOK - 1);   // clamp; pad rows discarded

  // Q fragment: real K=16 in v=0..3; rest zero at compile time.
  bf16x16 qf;
  {
    const float* qp = Q + ((size_t)b*TOK + ar)*Cc + h*DHd + half*8;
#pragma unroll
    for (int v = 0; v < 4; ++v) {
      const float2 qv = *(const float2*)(qp + v*2);
      qf[2*v]   = (__bf16)(qv.x * 0.25f);     // 1/sqrt(DH)
      qf[2*v+1] = (__bf16)(qv.y * 0.25f);
    }
#pragma unroll
    for (int v = 4; v < 8; ++v) { qf[2*v] = (__bf16)0.f; qf[2*v+1] = (__bf16)0.f; }
  }

  for (int nt = 0; nt < Nn/16; ++nt) {
    const int n0 = nt*16;
    const bf16x16 kv = *(const bf16x16*)(Kbf + ((size_t)b*Nn + n0 + l16)*Cc + h*DHd);
    bf16x16 kf;
#pragma unroll
    for (int e = 0; e < 16; ++e) kf[e] = (e < 8) ? kv[half*8 + e] : (__bf16)0.f;
    f32x8 s = {};
    s = __builtin_amdgcn_wmma_f32_16x16x32_bf16(false, qf, false, kf,
                                                (short)0, s, false, false);
    const int key = n0 + l16;
    const bool mk = mask[(size_t)b*Nn + key] != 0;
#pragma unroll
    for (int v = 0; v < 8; ++v)
      sS[(half*8 + v)*Nn + key] = mk ? -1e9f : s[v];
  }
  __syncthreads();

  // softmax per row; final probs go to LDS as row-major bf16 (unswizzled)
  for (int r = 0; r < 16; ++r) {
    float mx = -1e30f;
    for (int j = lane; j < Nn; j += 32) mx = fmaxf(mx, sS[r*Nn + j]);
    for (int o = 16; o > 0; o >>= 1) mx = fmaxf(mx, __shfl_xor(mx, o, 32));
    float sum = 0.f;
    for (int j = lane; j < Nn; j += 32) {
      float e = expf(sS[r*Nn + j] - mx);
      sS[r*Nn + j] = e; sum += e;
    }
    for (int o = 16; o > 0; o >>= 1) sum += __shfl_xor(sum, o, 32);
    const float inv = 1.0f / sum;
    for (int j = lane; j < Nn; j += 32)
      sP[r*Nn + j] = (__bf16)(sS[r*Nn + j] * inv);
  }
  __syncthreads();

  // O = P @ V : both fragments are single contiguous vector loads
  f32x8 acc = {};
  const __bf16* vr = Vtb + ((size_t)b*Cc + h*DHd + l16)*Nn + half*16;
  const __bf16* pr = sP + (size_t)l16*Nn + half*16;
  for (int kt = 0; kt < Nn/32; ++kt) {
    const int k0 = kt*32;
    const bf16x16 vf = *(const bf16x16*)(vr + k0);
    const bf16x16 pf = *(const bf16x16*)(pr + k0);
    acc = __builtin_amdgcn_wmma_f32_16x16x32_bf16(false, pf, false, vf,
                                                  (short)0, acc, false, false);
  }
  const int sc = swz(h*DHd + l16);
#pragma unroll
  for (int v = 0; v < 8; ++v) {
    int m = m0 + half*8 + v;
    if (m < TOK) Ob[((size_t)b*TOK + m)*Cc + sc] = (__bf16)acc[v];
  }
}

// ========= LayerNorm: f32 in -> bf16-swizzled out (feeds GEMMs only) =========
__global__ __launch_bounds__(256)
void layernorm_k(const float* __restrict__ X, const float* __restrict__ w,
                 const float* __restrict__ bia, __bf16* __restrict__ Yb, int tokens)
{
  int wid = threadIdx.x >> 5, lane = threadIdx.x & 31;
  int tok = blockIdx.x * 8 + wid;
  if (tok >= tokens) return;
  const float4 xv = ((const float4*)(X + (size_t)tok*Cc))[lane];
  float s = xv.x + xv.y + xv.z + xv.w;
  for (int o = 16; o > 0; o >>= 1) s += __shfl_xor(s, o, 32);
  float mean = s * (1.0f/Cc);
  float dx = xv.x - mean, dy = xv.y - mean, dz = xv.z - mean, dw = xv.w - mean;
  float v2 = dx*dx + dy*dy + dz*dz + dw*dw;
  for (int o = 16; o > 0; o >>= 1) v2 += __shfl_xor(v2, o, 32);
  float rstd = rsqrtf(v2*(1.0f/Cc) + 1e-5f);
  const float4 wv = ((const float4*)w)[lane];
  const float4 bv = ((const float4*)bia)[lane];
  __bf16* yp = Yb + (size_t)tok*Cc + swz(lane*4);  // 4 consecutive (bits 0-1 free)
  yp[0] = (__bf16)(dx*rstd*wv.x + bv.x);
  yp[1] = (__bf16)(dy*rstd*wv.y + bv.y);
  yp[2] = (__bf16)(dz*rstd*wv.z + bv.z);
  yp[3] = (__bf16)(dw*rstd*wv.w + bv.w);
}

// ===== RMSNorm w/ residual bookkeeping; optional f32 and bf16-swz outputs =====
// mode 0: r=x, Resid=r ; mode 1: r=x+Resid, Resid=r ; mode 2: r=x+Resid (no writeback)
__global__ __launch_bounds__(256)
void rmsnorm_k(const float* __restrict__ X, float* __restrict__ Resid,
               const float* __restrict__ w, float* __restrict__ H,
               __bf16* __restrict__ Hb, int tokens, int mode)
{
  int wid = threadIdx.x >> 5, lane = threadIdx.x & 31;
  int tok = blockIdx.x * 8 + wid;
  if (tok >= tokens) return;
  float4 xv = ((const float4*)(X + (size_t)tok*Cc))[lane];
  if (mode != 0) {
    float4 rv = ((const float4*)(Resid + (size_t)tok*Cc))[lane];
    xv.x += rv.x; xv.y += rv.y; xv.z += rv.z; xv.w += rv.w;
  }
  if (mode != 2) ((float4*)(Resid + (size_t)tok*Cc))[lane] = xv;
  float v2 = xv.x*xv.x + xv.y*xv.y + xv.z*xv.z + xv.w*xv.w;
  for (int o = 16; o > 0; o >>= 1) v2 += __shfl_xor(v2, o, 32);
  float rms = rsqrtf(v2*(1.0f/Cc) + 1e-5f);
  const float4 wv = ((const float4*)w)[lane];
  float4 ov;
  ov.x = xv.x*rms*wv.x; ov.y = xv.y*rms*wv.y;
  ov.z = xv.z*rms*wv.z; ov.w = xv.w*rms*wv.w;
  if (H) ((float4*)(H + (size_t)tok*Cc))[lane] = ov;
  if (Hb) {
    __bf16* yp = Hb + (size_t)tok*Cc + swz(lane*4);
    yp[0] = (__bf16)ov.x; yp[1] = (__bf16)ov.y;
    yp[2] = (__bf16)ov.z; yp[3] = (__bf16)ov.w;
  }
}

// ===== Causal depthwise conv (DC=4) + SiLU; dual f32 + bf16-swz output =====
__global__ __launch_bounds__(256)
void conv_silu_k(const float* __restrict__ UZ, const float* __restrict__ cw,
                 const float* __restrict__ cb, float* __restrict__ UC,
                 __bf16* __restrict__ UCb, int dir)
{
  int gid = blockIdx.x * blockDim.x + threadIdx.x;
  if (gid >= SEQ*Tt*DIi) return;
  int di  = gid & (DIi - 1);
  int rem = gid >> 8;
  int tau = rem % Tt;
  int seq = rem / Tt;
  float a = cb[di];
#pragma unroll
  for (int k = 0; k < DCc; ++k) {
    int p = tau - (DCc - 1) + k;
    if (p >= 0) {
      int to = dir ? (Tt - 1 - p) : p;
      a += UZ[((size_t)(seq*Tt + to))*(2*DIi) + di] * cw[di*DCc + k];
    }
  }
  float v = silu_f(a);
  UC[(size_t)gid] = v;
  UCb[(size_t)(gid & ~(DIi-1)) + swz(di)] = (__bf16)v;
}

// ===== Selective scan: one block per sequence, one thread per DI channel =====
__global__ __launch_bounds__(256)
void ssm_scan_k(const float* __restrict__ UC, const float* __restrict__ XD,
                const float* __restrict__ DT, const float* __restrict__ Alog,
                const float* __restrict__ Dp, float* __restrict__ Y, int dir)
{
  __shared__ float sB[Tt * DSs];
  __shared__ float sC[Tt * DSs];
  const int seq = blockIdx.x;
  const int di  = threadIdx.x;
  for (int idx = di; idx < Tt*32; idx += 256) {
    int tau = idx >> 5, j = idx & 31;
    float val = XD[((size_t)(seq*Tt + tau))*40 + DRr + j];
    if (j < DSs) sB[tau*DSs + j] = val;
    else         sC[tau*DSs + (j - DSs)] = val;
  }
  __syncthreads();
  float a[DSs], hst[DSs];
#pragma unroll
  for (int s = 0; s < DSs; ++s) { a[s] = -expf(Alog[di*DSs + s]); hst[s] = 0.f; }
  const float Dv = Dp[di];
  for (int tau = 0; tau < Tt; ++tau) {
    size_t row = (size_t)(seq*Tt + tau);
    float dtv = DT[row*DIi + di];
    float ucv = UC[row*DIi + di];
    float du  = dtv * ucv;
    float y   = 0.f;
#pragma unroll
    for (int s = 0; s < DSs; ++s) {
      hst[s] = expf(dtv*a[s])*hst[s] + du*sB[tau*DSs + s];
      y += hst[s]*sC[tau*DSs + s];
    }
    y += ucv * Dv;
    int to = dir ? (Tt - 1 - tau) : tau;
    Y[((size_t)(seq*Tt + to))*DIi + di] = y;
  }
}

// ===== gating: yc = (yf + yb) * silu(z) -> bf16-swz (feeds out-proj GEMM) =====
__global__ __launch_bounds__(256)
void gate_k(const float* __restrict__ yf, const float* __restrict__ yb,
            const float* __restrict__ UZ, __bf16* __restrict__ YCb)
{
  int gid = blockIdx.x * blockDim.x + threadIdx.x;
  if (gid >= ROWS*DIi) return;
  int di  = gid & (DIi - 1);
  int row = gid >> 8;
  float z = UZ[(size_t)row*(2*DIi) + DIi + di];
  YCb[(size_t)(gid & ~(DIi-1)) + swz(di)] = (__bf16)((yf[gid] + yb[gid]) * silu_f(z));
}

// ===== max over T -> bf16-swz (feeds score-head GEMM) =====
__global__ __launch_bounds__(256)
void maxt_k(const float* __restrict__ X, __bf16* __restrict__ XMb)
{
  int gid = blockIdx.x * blockDim.x + threadIdx.x;
  if (gid >= SEQ*Cc) return;
  int c  = gid & (Cc - 1);
  int bm = gid >> 7;
  float mx = -1e30f;
  for (int t = 0; t < Tt; ++t)
    mx = fmaxf(mx, X[((size_t)bm*Tt + t)*Cc + c]);
  XMb[(size_t)bm*Cc + swz(c)] = (__bf16)mx;
}

// ===== small head: out = post(hid[64] @ w2[64,nOut] + b2) ; post 1 = elu+1.0001 =====
__global__ __launch_bounds__(256)
void head_out_k(const float* __restrict__ hid, const float* __restrict__ w2,
                const float* __restrict__ b2, float* __restrict__ out,
                int rows, int nOut, int post)
{
  int row = blockIdx.x * blockDim.x + threadIdx.x;
  if (row >= rows) return;
  for (int n = 0; n < nOut; ++n) {
    float s = b2[n];
    for (int k = 0; k < 64; ++k) s += hid[(size_t)row*64 + k] * w2[k*nOut + n];
    if (post == 1) s = ((s > 0.f) ? s : (expf(s) - 1.f)) + 1.0001f;
    out[(size_t)row*nOut + n] = s;
  }
}

// =====================================================================
extern "C" void kernel_launch(void* const* d_in, const int* in_sizes, int n_in,
                              void* d_out, int out_size, void* d_ws, size_t ws_size,
                              hipStream_t stream)
{
  (void)in_sizes; (void)n_in; (void)out_size; (void)ws_size;
  const float* mode_in  = (const float*)d_in[0];
  const float* enc      = (const float*)d_in[1];
  const unsigned char* mask = (const unsigned char*)d_in[2];
  const float* ca_ln1_w = (const float*)d_in[3];
  const float* ca_ln1_b = (const float*)d_in[4];
  const float* ca_wq = (const float*)d_in[5];
  const float* ca_wk = (const float*)d_in[6];
  const float* ca_wv = (const float*)d_in[7];
  const float* ca_wo = (const float*)d_in[8];
  const float* ca_bq = (const float*)d_in[9];
  const float* ca_bk = (const float*)d_in[10];
  const float* ca_bv = (const float*)d_in[11];
  const float* ca_bo = (const float*)d_in[12];
  const float* ca_ln2_w = (const float*)d_in[13];
  const float* ca_ln2_b = (const float*)d_in[14];
  const float* ca_w1 = (const float*)d_in[15];
  const float* ca_b1 = (const float*)d_in[16];
  const float* ca_w2 = (const float*)d_in[17];
  const float* ca_b2 = (const float*)d_in[18];
  const float* mb_norm_w  = (const float*)d_in[19];
  const float* mb_in_w    = (const float*)d_in[20];
  const float* mb_conv_w  = (const float*)d_in[21];
  const float* mb_conv_b  = (const float*)d_in[22];
  const float* mb_xproj_w = (const float*)d_in[23];
  const float* mb_dt_w    = (const float*)d_in[24];
  const float* mb_dt_b    = (const float*)d_in[25];
  const float* mb_Alog    = (const float*)d_in[26];
  const float* mb_D       = (const float*)d_in[27];
  const float* mb_out_w   = (const float*)d_in[28];
  const float* fnorm_w    = (const float*)d_in[29];
  const float* pr_g1 = (const float*)d_in[30];
  const float* pr_gb1= (const float*)d_in[31];
  const float* pr_g2 = (const float*)d_in[32];
  const float* pr_gb2= (const float*)d_in[33];
  const float* pr_s1 = (const float*)d_in[34];
  const float* pr_sb1= (const float*)d_in[35];
  const float* pr_s2 = (const float*)d_in[36];
  const float* pr_sb2= (const float*)d_in[37];
  const float* pr_c1 = (const float*)d_in[38];
  const float* pr_cb1= (const float*)d_in[39];
  const float* pr_c2 = (const float*)d_in[40];
  const float* pr_cb2= (const float*)d_in[41];

  float* ws = (float*)d_ws;
  float* outp = (float*)d_out;

  // ---- f32 workspace (floats) ----
  const size_t SZ_X  = (size_t)ROWS * Cc;
  const size_t SZ_KV = (size_t)ENCR * Cc;
  const size_t SZ_UZ = (size_t)ROWS * 2*DIi;
  const size_t SZ_DI = (size_t)ROWS * DIi;
  const size_t SZ_XD = (size_t)ROWS * 40;

  const size_t o_x    = 0;
  const size_t o_res  = o_x   + SZ_X;
  const size_t o_q    = o_res + SZ_X;          // attention phase (region reused)
  const size_t o_k    = o_q   + SZ_X;
  const size_t o_v    = o_k   + SZ_KV;
  const size_t o_qend = o_q + ((SZ_X + 2*SZ_KV > 2*SZ_DI) ? (SZ_X + 2*SZ_KV) : 2*SZ_DI);
  const size_t o_ucf  = o_q;                   // alias: q/k/v dead during mamba
  const size_t o_ucb  = o_q + SZ_DI;
  const size_t o_hid  = o_q;                   // alias: predictor phase
  const size_t o_hs   = o_q + (size_t)ROWS*64;
  const size_t o_uz   = o_qend;
  const size_t o_xdf  = o_uz  + SZ_UZ;
  const size_t o_xdb  = o_xdf + SZ_XD;
  const size_t o_dtf  = o_xdb + SZ_XD;
  const size_t o_dtb  = o_dtf + SZ_DI;
  const size_t o_yf   = o_dtb + SZ_DI;
  const size_t o_yb   = o_yf  + SZ_DI;

  // ---- bf16 workspace (elements), 32B aligned ----
  const size_t o_bf = (o_yb + SZ_DI + 7) & ~(size_t)7;
  __bf16* bf = (__bf16*)(ws + o_bf);
  const size_t LCC = (size_t)Llay*Cc*Cc;
  const size_t b_wq  = 0;
  const size_t b_wk  = b_wq  + LCC;
  const size_t b_wv  = b_wk  + LCC;
  const size_t b_wo  = b_wv  + LCC;
  const size_t b_w1  = b_wo  + LCC;                    // [L][FF][C]
  const size_t b_w2  = b_w1  + (size_t)Llay*Cc*FFf;    // [L][C][FF]
  const size_t b_in  = b_w2  + (size_t)Llay*FFf*Cc;    // [L][2DI][C]
  const size_t b_ow  = b_in  + (size_t)Llay*Cc*2*DIi;  // [L][C][DI]
  const size_t b_xp  = b_ow  + (size_t)Llay*DIi*Cc;    // [L*2][40][DI]
  const size_t b_g1  = b_xp  + (size_t)Llay*2*DIi*40;  // [2][64][C]
  const size_t b_c1  = b_g1  + (size_t)2*Cc*64;
  const size_t b_s1  = b_c1  + (size_t)2*Cc*64;
  const size_t b_eb  = b_s1  + (size_t)2*Cc*64;        // enc bf16-swz
  const size_t b_kbf = b_eb  + SZ_KV;
  const size_t b_vtb = b_kbf + SZ_KV;
  const size_t b_hl  = b_vtb + SZ_KV;                  // norm outs (bf16-swz)
  const size_t b_ff  = b_hl  + SZ_X;                   // ff1 out
  const size_t b_at  = b_ff  + (size_t)ROWS*FFf;       // attn out
  const size_t b_ucf = b_at  + SZ_X;                   // conv outs
  const size_t b_ucb = b_ucf + SZ_DI;
  const size_t b_yc  = b_ucb + SZ_DI;                  // gate out
  const size_t b_pb  = b_yc  + SZ_DI;                  // stage-out bf16 copy
  const size_t b_xm  = b_pb  + SZ_X;                   // maxT out

  // ---- output layout (floats, reference return order) ----
  const size_t OO_YHAT  = 0;
  const size_t OO_PI    = OO_YHAT + (size_t)Bb*Mm*Tt*2;
  const size_t OO_MODE2 = OO_PI   + (size_t)Bb*Mm;
  const size_t OO_YHATN = OO_MODE2+ SZ_X;
  const size_t OO_PIN   = OO_YHATN+ (size_t)Bb*Mm*Tt*2;
  const size_t OO_SCAL  = OO_PIN  + (size_t)Bb*Mm;
  const size_t OO_SCALN = OO_SCAL + (size_t)Bb*Mm*Tt*2;

  auto WT = [&](const float* W, size_t boff, int Lm, int K, int N) {
    int total = Lm*K*N;
    wconv_k<<<(total + 255)/256, 256, 0, stream>>>(W, bf + boff, total, K, N);
  };
  // rows%64==0, K%32==0 for all callers
  auto GB = [&](size_t aoff, size_t boff, const float* bias, const float* resid,
                float* out, __bf16* outb, int ldo, int rows, int K, int N, int act) {
    dim3 g(rows/64, (N + 15)/16);
    gemm_bb_k<<<g, 32, 0, stream>>>(bf + aoff, bf + boff, bias, resid, out, outb,
                                    ldo, K, N, act);
  };

  // ---- one-time packs (graph-capture safe, on stream) ----
  WT(ca_wq, b_wq, Llay, Cc, Cc);
  WT(ca_wk, b_wk, Llay, Cc, Cc);
  WT(ca_wv, b_wv, Llay, Cc, Cc);
  WT(ca_wo, b_wo, Llay, Cc, Cc);
  WT(ca_w1, b_w1, Llay, Cc, FFf);
  WT(ca_w2, b_w2, Llay, FFf, Cc);
  WT(mb_in_w, b_in, Llay, Cc, 2*DIi);
  WT(mb_out_w, b_ow, Llay, DIi, Cc);
  WT(mb_xproj_w, b_xp, Llay*2, DIi, DRr + 2*DSs);
  WT(pr_g1, b_g1, 2, Cc, 64);
  WT(pr_c1, b_c1, 2, Cc, 64);
  WT(pr_s1, b_s1, 2, Cc, 64);
  encpack_k<<<(ENCR*Cc + 255)/256, 256, 0, stream>>>(enc, bf + b_eb);

  hipMemcpyAsync(ws + o_x, mode_in, SZ_X*sizeof(float),
                 hipMemcpyDeviceToDevice, stream);

  float* x  = ws + o_x;
  float* rs = ws + o_res;

  for (int stage = 0; stage < 2; ++stage) {
    int off = stage ? L1l : 0;
    int nl  = stage ? L2l : L1l;
    for (int li = 0; li < nl; ++li) {
      int i = off + li;
      // ---------- cross block ----------
      layernorm_k<<<(ROWS+7)/8, 256, 0, stream>>>(x, ca_ln1_w + i*Cc, ca_ln1_b + i*Cc, bf + b_hl, ROWS);
      GB(b_hl, b_wq + (size_t)i*Cc*Cc, ca_bq + i*Cc, nullptr, ws + o_q, nullptr, Cc, ROWS, Cc, Cc, 0);
      GB(b_eb, b_wk + (size_t)i*Cc*Cc, ca_bk + i*Cc, nullptr, ws + o_k, nullptr, Cc, ENCR, Cc, Cc, 0);
      GB(b_eb, b_wv + (size_t)i*Cc*Cc, ca_bv + i*Cc, nullptr, ws + o_v, nullptr, Cc, ENCR, Cc, Cc, 0);
      kvpack_k<<<(ENCR*Cc + 255)/256, 256, 0, stream>>>(ws + o_k, ws + o_v, bf + b_kbf, bf + b_vtb);
      attn_wmma_k<<<dim3((TOK+15)/16, Hh, Bb), 32, 0, stream>>>(ws + o_q, bf + b_kbf, bf + b_vtb, mask, bf + b_at);
      GB(b_at, b_wo + (size_t)i*Cc*Cc, ca_bo + i*Cc, x, x, nullptr, Cc, ROWS, Cc, Cc, 0);
      layernorm_k<<<(ROWS+7)/8, 256, 0, stream>>>(x, ca_ln2_w + i*Cc, ca_ln2_b + i*Cc, bf + b_hl, ROWS);
      GB(b_hl, b_w1 + (size_t)i*Cc*FFf, ca_b1 + i*FFf, nullptr, nullptr, bf + b_ff, FFf, ROWS, Cc, FFf, 1);
      GB(b_ff, b_w2 + (size_t)i*FFf*Cc, ca_b2 + i*Cc, x, x, nullptr, Cc, ROWS, FFf, Cc, 0);
      // ---------- mamba block ----------
      rmsnorm_k<<<(ROWS+7)/8, 256, 0, stream>>>(x, rs, mb_norm_w + i*Cc, nullptr, bf + b_hl, ROWS, li == 0 ? 0 : 1);
      GB(b_hl, b_in + (size_t)i*Cc*2*DIi, nullptr, nullptr, ws + o_uz, nullptr, 2*DIi, ROWS, Cc, 2*DIi, 0);
      for (int d = 0; d < 2; ++d) {
        const float* cw = mb_conv_w  + ((size_t)i*2 + d)*DIi*DCc;
        const float* cb = mb_conv_b  + ((size_t)i*2 + d)*DIi;
        const float* dw = mb_dt_w    + ((size_t)i*2 + d)*DRr*DIi;
        const float* db = mb_dt_b    + ((size_t)i*2 + d)*DIi;
        const float* al = mb_Alog    + ((size_t)i*2 + d)*DIi*DSs;
        const float* Dd = mb_D       + ((size_t)i*2 + d)*DIi;
        float* uc = ws + (d ? o_ucb : o_ucf);
        size_t ucbf = d ? b_ucb : b_ucf;
        float* xd = ws + (d ? o_xdb : o_xdf);
        float* dt = ws + (d ? o_dtb : o_dtf);
        float* yy = ws + (d ? o_yb  : o_yf);
        conv_silu_k<<<(SEQ*Tt*DIi + 255)/256, 256, 0, stream>>>(ws + o_uz, cw, cb, uc, bf + ucbf, d);
        GB(ucbf, b_xp + ((size_t)i*2 + d)*DIi*40, nullptr, nullptr, xd, nullptr, 40, ROWS, DIi, 40, 0);
        gemm_edge_k<<<dim3(ROWS/16, DIi/16), 32, 0, stream>>>(xd, 40, dw, DIi, db, dt, DIi, DRr, DIi, 2);
        ssm_scan_k<<<SEQ, 256, 0, stream>>>(uc, xd, dt, al, Dd, yy, d);
      }
      gate_k<<<(ROWS*DIi + 255)/256, 256, 0, stream>>>(ws + o_yf, ws + o_yb, ws + o_uz, bf + b_yc);
      GB(b_yc, b_ow + (size_t)i*DIi*Cc, nullptr, nullptr, x, nullptr, Cc, ROWS, DIi, Cc, 0);
    }
    // stage final: rmsnorm(x + residual) * fnorm_w[stage]; dual f32 + bf16 outputs
    float* dst = stage ? (outp + OO_MODE2) : x;
    rmsnorm_k<<<(ROWS+7)/8, 256, 0, stream>>>(x, rs, fnorm_w + stage*Cc, dst, bf + b_pb, ROWS, 2);

    // ---------- predictor on this stage's output ----------
    float* hid = ws + o_hid;
    GB(b_pb, b_g1 + (size_t)stage*Cc*64, pr_gb1 + stage*64, nullptr, hid, nullptr, 64, ROWS, Cc, 64, 1);
    head_out_k<<<(ROWS+255)/256, 256, 0, stream>>>(hid, pr_g2 + (size_t)stage*64*2,
        pr_gb2 + stage*2, outp + (stage ? OO_YHATN : OO_YHAT), ROWS, 2, 0);
    GB(b_pb, b_c1 + (size_t)stage*Cc*64, pr_cb1 + stage*64, nullptr, hid, nullptr, 64, ROWS, Cc, 64, 1);
    head_out_k<<<(ROWS+255)/256, 256, 0, stream>>>(hid, pr_c2 + (size_t)stage*64*2,
        pr_cb2 + stage*2, outp + (stage ? OO_SCALN : OO_SCAL), ROWS, 2, 1);
    maxt_k<<<(SEQ*Cc + 255)/256, 256, 0, stream>>>(dst, bf + b_xm);
    GB(b_xm, b_s1 + (size_t)stage*Cc*64, pr_sb1 + stage*64, nullptr, ws + o_hs, nullptr, 64, SEQ, Cc, 64, 1);
    head_out_k<<<(SEQ+255)/256, 256, 0, stream>>>(ws + o_hs, pr_s2 + (size_t)stage*64,
        pr_sb2 + stage, outp + (stage ? OO_PIN : OO_PI), SEQ, 1, 0);
  }
}
